// GPS_15479062135042
// MI455X (gfx1250) — matrix operand order repaired and checked by reading, not compiled, not run on hardware
//
#include <hip/hip_runtime.h>
#include <hip/hip_bf16.h>
#include <math.h>

// ---------------------------------------------------------------------------
// GraphGPS forward for MI455X (gfx1250).
//  - All GEMMs: v_wmma_f32_16x16x32_f16 (f16 in, f32 accumulate)
//  - All tile staging: async LDS DMA (global_load_async_to_lds_b128,
//    ASYNCcnt) with double-buffered software pipelining.
// ---------------------------------------------------------------------------

#define GG 32
#define NPER 512
#define CC 256
#define HH 8
#define HD 32
#define LL 4
#define NN (GG * NPER)          // 16384
#define EE (NN * 16)            // 262144 edges
#define ECHUNK 16384
#define NCHUNKS (EE / ECHUNK)   // 16

typedef _Float16 half_t;
typedef _Float16 v16h __attribute__((ext_vector_type(16)));
typedef float    v8f  __attribute__((ext_vector_type(8)));

union frag16 { v16h v; unsigned int u[8]; };
union hpack  { unsigned int u; half_t h[2]; };

#define ASYNC_CP_B128(ldsoff, gaddr)                                        \
  asm volatile("global_load_async_to_lds_b128 %0, %1, off"                  \
               :: "v"(ldsoff), "v"(gaddr) : "memory")

// ---------------------------------------------------------------------------
// GEMM: D[M,Nc](f32) = act(A[M,K](f16) * W[K,Nc](f16) + bias(f32)).
// Block = 128 threads (4 waves); block tile 64x64; wave tile 32x32
// (2x2 WMMA tiles); K-step 32. Interior tiles: async-DMA staging,
// double buffered. Edge tiles: branchless clamped scalar staging.
// ---------------------------------------------------------------------------
#define SA 40   // A LDS row stride (elems) : 80B rows, 16B-aligned segments
#define SB 80   // B LDS row stride (elems) : 160B rows, 16B-aligned segments
__global__ __launch_bounds__(128) void k_gemm(
    const half_t* __restrict__ A, int lda,
    const half_t* __restrict__ W, int ldw,
    const float* __restrict__ bias,
    float* __restrict__ D, int ldd,
    int M, int K, int Nc, int relu)
{
  __shared__ half_t As[2][64 * SA];   // [row][k] natural, f16
  __shared__ half_t Bs[2][32 * SB];   // [k][n]  natural, f16

  const int tid  = threadIdx.x;
  const int lane = tid & 31;
  const int wave = tid >> 5;
  const int bm = blockIdx.y * 64;
  const int bn = blockIdx.x * 64;
  const int wm = (wave >> 1) * 32;
  const int wn = (wave & 1) * 32;
  const int m16  = lane & 15;
  const int half = lane >> 4;
  const int nk = (K + 31) >> 5;
  const bool fastAll = (bm + 64 <= M) && (bn + 64 <= Nc) &&
                       ((K & 31) == 0) && ((lda & 7) == 0) && ((ldw & 7) == 0);

  v8f acc[2][2];
  #pragma unroll
  for (int i = 0; i < 2; ++i)
    #pragma unroll
    for (int j = 0; j < 2; ++j)
      #pragma unroll
      for (int e = 0; e < 8; ++e) acc[i][j][e] = 0.0f;

  // async staging: A tile = 256 16B segs, B tile = 256 16B segs -> 4/thread
  auto stage_fast = [&](int buf, int k0) {
    #pragma unroll
    for (int it = 0; it < 2; ++it) {
      int idx = tid + it * 128;
      int r = idx >> 2, sg = idx & 3;
      unsigned lds = (unsigned)(size_t)&As[buf][r * SA + sg * 8];
      unsigned long long ga =
          (unsigned long long)(size_t)&A[(size_t)(bm + r) * lda + k0 + sg * 8];
      ASYNC_CP_B128(lds, ga);
    }
    #pragma unroll
    for (int it = 0; it < 2; ++it) {
      int idx = tid + it * 128;
      int kk = idx >> 3, sg = idx & 7;
      unsigned lds = (unsigned)(size_t)&Bs[buf][kk * SB + sg * 8];
      unsigned long long gb =
          (unsigned long long)(size_t)&W[(size_t)(k0 + kk) * ldw + bn + sg * 8];
      ASYNC_CP_B128(lds, gb);
    }
  };

  auto stage_slow = [&](int buf, int k0) {
    #pragma unroll
    for (int it = 0; it < 16; ++it) {
      int i = tid + it * 128;
      int r = i >> 5, c = i & 31;
      int gr = bm + r, gc = k0 + c;
      int cr = gr < M ? gr : M - 1;
      int cc = gc < K ? gc : K - 1;
      half_t v = A[(size_t)cr * lda + cc];
      As[buf][r * SA + c] = (gr < M && gc < K) ? v : (half_t)0.0f;
    }
    #pragma unroll
    for (int it = 0; it < 16; ++it) {
      int i = tid + it * 128;
      int kk = i >> 6, nn = i & 63;
      int gk = k0 + kk, gn = bn + nn;
      int ck = gk < K ? gk : K - 1;
      int cn = gn < Nc ? gn : Nc - 1;
      half_t v = W[(size_t)ck * ldw + cn];
      Bs[buf][kk * SB + nn] = (gk < K && gn < Nc) ? v : (half_t)0.0f;
    }
  };

  auto compute = [&](int buf) {
    frag16 af[2], bf[2];
    // A frag: lane m=lane&15, element i -> K = g*16 + half*8 + (i&7)
    #pragma unroll
    for (int ti = 0; ti < 2; ++ti) {
      const half_t* ap = &As[buf][(wm + ti * 16 + m16) * SA];
      #pragma unroll
      for (int g2 = 0; g2 < 2; ++g2)
        #pragma unroll
        for (int j = 0; j < 4; ++j)
          af[ti].u[g2 * 4 + j] =
              *(const unsigned int*)&ap[g2 * 16 + half * 8 + 2 * j];
    }
    // B frag from natural [k][n]: element i -> K = (i/8)*16 + half*8 + (i&7) (u16 gathers)
    #pragma unroll
    for (int tj = 0; tj < 2; ++tj) {
      int ncol = wn + tj * 16 + m16;
      #pragma unroll
      for (int q = 0; q < 8; ++q) {
        hpack hp;
        hp.h[0] = Bs[buf][((q >> 2) * 16 + half * 8 + 2 * (q & 3) + 0) * SB + ncol];
        hp.h[1] = Bs[buf][((q >> 2) * 16 + half * 8 + 2 * (q & 3) + 1) * SB + ncol];
        bf[tj].u[q] = hp.u;
      }
    }
    #pragma unroll
    for (int ti = 0; ti < 2; ++ti)
      #pragma unroll
      for (int tj = 0; tj < 2; ++tj)
        acc[ti][tj] = __builtin_amdgcn_wmma_f32_16x16x32_f16(
            false, af[ti].v, false, bf[tj].v, (short)0, acc[ti][tj],
            false, false);
  };

  if (fastAll) {
    stage_fast(0, 0);
    int cur = 0;
    for (int ks = 0; ks < nk; ++ks) {
      if (ks + 1 < nk) {
        stage_fast(cur ^ 1, (ks + 1) * 32);
        asm volatile("s_wait_asynccnt 0x4" ::: "memory");
      } else {
        asm volatile("s_wait_asynccnt 0x0" ::: "memory");
      }
      __syncthreads();
      compute(cur);
      __syncthreads();
      cur ^= 1;
    }
  } else {
    for (int ks = 0; ks < nk; ++ks) {
      stage_slow(0, ks * 32);
      __syncthreads();
      compute(0);
      __syncthreads();
    }
  }

  // epilogue: C/D layout -> lane holds col n=lane&15, rows r + 8*half
  #pragma unroll
  for (int tj = 0; tj < 2; ++tj) {
    int col = bn + wn + tj * 16 + m16;
    if (col >= Nc) continue;
    float bv = bias ? bias[col] : 0.0f;
    #pragma unroll
    for (int ti = 0; ti < 2; ++ti) {
      #pragma unroll
      for (int r = 0; r < 8; ++r) {
        int row = bm + wm + ti * 16 + half * 8 + r;
        if (row < M) {
          float v = acc[ti][tj][r] + bv;
          if (relu) v = fmaxf(v, 0.0f);
          D[(size_t)row * ldd + col] = v;
        }
      }
    }
  }
}

// f32 -> f16 convert (feeds async-DMA staging)
__global__ void k_cvt_h(const float* __restrict__ in, half_t* __restrict__ out,
                        size_t n)
{
  for (size_t i = blockIdx.x * (size_t)blockDim.x + threadIdx.x; i < n;
       i += (size_t)gridDim.x * blockDim.x)
    out[i] = (half_t)in[i];
}

// ---------------------------------------------------------------------------
// Flash attention per (graph, head): S=512, HD=32. Block = 128 threads,
// 4 waves; each wave owns a 16-row Q tile. qkvh is f16 [N, 768] (Q|K|V).
// K/V chunks staged via double-buffered async LDS DMA. grid = (S/64, G*H).
// ---------------------------------------------------------------------------
#define ST 40
__global__ __launch_bounds__(128) void k_attn(
    const half_t* __restrict__ qkvh, float* __restrict__ o)
{
  __shared__ half_t Ks[2][32 * ST];     // [key][hd]
  __shared__ half_t Vs[2][32 * ST];     // [key][hd]
  __shared__ half_t Pb[4][16 * ST];     // per-wave P tile [row][key]

  const int tid = threadIdx.x, lane = tid & 31, wave = tid >> 5;
  const int m16 = lane & 15, half = lane >> 4;
  const int gh = blockIdx.y, g = gh >> 3, hh = gh & 7;
  const int grow0 = g * NPER + blockIdx.x * 64 + wave * 16;
  const float qs = 0.17677669529663687f;  // 1/sqrt(32)

  // Q fragment straight from global f16 (pairs contiguous along hd)
  frag16 qf;
  #pragma unroll
  for (int g2 = 0; g2 < 2; ++g2)
    #pragma unroll
    for (int j = 0; j < 4; ++j) {
      int hd = g2 * 16 + half * 8 + 2 * j;
      qf.u[g2 * 4 + j] = *(const unsigned int*)
          &qkvh[(size_t)(grow0 + m16) * 768 + hh * 32 + hd];
    }

  const int skey = tid >> 2, sseg = tid & 3;
  auto issueKV = [&](int buf, int kc) {
    size_t base = (size_t)(g * NPER + kc * 32 + skey) * 768 +
                  hh * 32 + sseg * 8;
    unsigned lK = (unsigned)(size_t)&Ks[buf][skey * ST + sseg * 8];
    unsigned lV = (unsigned)(size_t)&Vs[buf][skey * ST + sseg * 8];
    unsigned long long gK = (unsigned long long)(size_t)&qkvh[base + 256];
    unsigned long long gV = (unsigned long long)(size_t)&qkvh[base + 512];
    ASYNC_CP_B128(lK, gK);
    ASYNC_CP_B128(lV, gV);
  };

  float mrow[8], lrow[8];
  v8f o0, o1;
  #pragma unroll
  for (int r = 0; r < 8; ++r) {
    mrow[r] = -1e30f; lrow[r] = 0.0f; o0[r] = 0.0f; o1[r] = 0.0f;
  }

  issueKV(0, 0);
  int cur = 0;
  for (int kc = 0; kc < NPER / 32; ++kc) {
    if (kc + 1 < NPER / 32) {
      issueKV(cur ^ 1, kc + 1);
      asm volatile("s_wait_asynccnt 0x2" ::: "memory");
    } else {
      asm volatile("s_wait_asynccnt 0x0" ::: "memory");
    }
    __syncthreads();

    frag16 kb[2];
    #pragma unroll
    for (int t = 0; t < 2; ++t) {
      const half_t* bp = &Ks[cur][(t * 16 + m16) * ST];
      #pragma unroll
      for (int q = 0; q < 8; ++q)
        kb[t].u[q] = *(const unsigned int*)&bp[(q >> 2) * 16 + half * 8 + 2 * (q & 3)];
    }
    v8f s0, s1;
    #pragma unroll
    for (int e = 0; e < 8; ++e) { s0[e] = 0.0f; s1[e] = 0.0f; }
    s0 = __builtin_amdgcn_wmma_f32_16x16x32_f16(false, qf.v, false, kb[0].v,
                                                (short)0, s0, false, false);
    s1 = __builtin_amdgcn_wmma_f32_16x16x32_f16(false, qf.v, false, kb[1].v,
                                                (short)0, s1, false, false);
    #pragma unroll
    for (int e = 0; e < 8; ++e) { s0[e] *= qs; s1[e] *= qs; }

    // online softmax; rows live as (r + 8*half) across 16-lane groups
    #pragma unroll
    for (int r = 0; r < 8; ++r) {
      float mx = fmaxf(s0[r], s1[r]);
      for (int off = 8; off; off >>= 1)
        mx = fmaxf(mx, __shfl_xor(mx, off, 16));
      float mn = fmaxf(mrow[r], mx);
      float f = __expf(mrow[r] - mn);
      float p0 = __expf(s0[r] - mn);
      float p1 = __expf(s1[r] - mn);
      float sr = p0 + p1;
      for (int off = 8; off; off >>= 1) sr += __shfl_xor(sr, off, 16);
      lrow[r] = lrow[r] * f + sr;
      mrow[r] = mn;
      o0[r] *= f; o1[r] *= f;
      int pr = r + 8 * half;
      Pb[wave][pr * ST + m16]      = (half_t)p0;
      Pb[wave][pr * ST + 16 + m16] = (half_t)p1;
    }

    // re-fragment P as A operand (intra-wave LDS, in-order)
    frag16 pf;
    {
      const half_t* pp = &Pb[wave][m16 * ST];
      #pragma unroll
      for (int g2 = 0; g2 < 2; ++g2)
        #pragma unroll
        for (int j = 0; j < 4; ++j)
          pf.u[g2 * 4 + j] =
              *(const unsigned int*)&pp[g2 * 16 + half * 8 + 2 * j];
    }
    // V fragments from natural [key][hd] layout: u16 gathers
    frag16 vb[2];
    #pragma unroll
    for (int t = 0; t < 2; ++t) {
      #pragma unroll
      for (int q = 0; q < 8; ++q) {
        hpack hp;
        hp.h[0] = Vs[cur][((q >> 2) * 16 + half * 8 + 2 * (q & 3) + 0) * ST + t * 16 + m16];
        hp.h[1] = Vs[cur][((q >> 2) * 16 + half * 8 + 2 * (q & 3) + 1) * ST + t * 16 + m16];
        vb[t].u[q] = hp.u;
      }
    }
    o0 = __builtin_amdgcn_wmma_f32_16x16x32_f16(false, pf.v, false, vb[0].v,
                                                (short)0, o0, false, false);
    o1 = __builtin_amdgcn_wmma_f32_16x16x32_f16(false, pf.v, false, vb[1].v,
                                                (short)0, o1, false, false);
    __syncthreads();
    cur ^= 1;
  }

  #pragma unroll
  for (int r = 0; r < 8; ++r) {
    float inv = 1.0f / lrow[r];
    int row = grow0 + r + 8 * half;
    o[(size_t)row * CC + hh * 32 + m16]      = o0[r] * inv;
    o[(size_t)row * CC + hh * 32 + 16 + m16] = o1[r] * inv;
  }
}

// ---------------------------------------------------------------------------
// GINE message + scatter: agg[dst] += relu(h[src] + ea). One wave per edge.
// ---------------------------------------------------------------------------
__global__ __launch_bounds__(256) void k_gine_scatter(
    const float* __restrict__ h, const float* __restrict__ ea,
    const int* __restrict__ src, const int* __restrict__ dst,
    float* __restrict__ agg, int nE)
{
  int e = (int)((blockIdx.x * (size_t)blockDim.x + threadIdx.x) >> 5);
  int lane = threadIdx.x & 31;
  if (e >= nE) return;
  int s = src[e], d = dst[e];
  #pragma unroll
  for (int j = 0; j < CC / 32; ++j) {
    int c = lane + j * 32;
    float m = h[(size_t)s * CC + c] + ea[(size_t)e * CC + c];
    m = fmaxf(m, 0.0f);
    atomicAdd(&agg[(size_t)d * CC + c], m);
  }
}

// out = a (+ b); stats[c] += sum, stats[C+c] += sumsq (out/b optional)
__global__ __launch_bounds__(256) void k_add_colstats(
    const float* __restrict__ a, const float* __restrict__ b,
    float* __restrict__ out, float* __restrict__ stats, int Nrows, int C)
{
  int c = blockIdx.x * 256 + threadIdx.x;
  if (c >= C) return;
  int r0 = blockIdx.y * 256;
  int r1 = r0 + 256; if (r1 > Nrows) r1 = Nrows;
  float s = 0.0f, s2 = 0.0f;
  for (int r = r0; r < r1; ++r) {
    float v = a[(size_t)r * C + c];
    if (b) v += b[(size_t)r * C + c];
    if (out) out[(size_t)r * C + c] = v;
    s += v; s2 += v * v;
  }
  atomicAdd(&stats[c], s);
  atomicAdd(&stats[C + c], s2);
}

__global__ void k_bn_apply(
    const float* __restrict__ in, float* __restrict__ out,
    const float* __restrict__ stats, const float* __restrict__ gamma,
    const float* __restrict__ beta, int Nrows, int C)
{
  size_t total = (size_t)Nrows * C;
  for (size_t i = blockIdx.x * (size_t)blockDim.x + threadIdx.x; i < total;
       i += (size_t)gridDim.x * blockDim.x) {
    int c = (int)(i % C);
    float mu  = stats[c] / (float)Nrows;
    float var = stats[C + c] / (float)Nrows - mu * mu;
    out[i] = (in[i] - mu) * rsqrtf(var + 1e-5f) * gamma[c] + beta[c];
  }
}

__global__ void k_add(const float* __restrict__ a, const float* __restrict__ b,
                      float* __restrict__ out, size_t n)
{
  for (size_t i = blockIdx.x * (size_t)blockDim.x + threadIdx.x; i < n;
       i += (size_t)gridDim.x * blockDim.x)
    out[i] = a[i] + b[i];
}

__global__ __launch_bounds__(256) void k_pool(const float* __restrict__ h,
                                              float* __restrict__ pooled)
{
  int g = blockIdx.x, c = threadIdx.x;
  float s = 0.0f;
  for (int r = 0; r < NPER; ++r) s += h[((size_t)g * NPER + r) * CC + c];
  pooled[g * CC + c] = s * (1.0f / NPER);
}

// ---------------------------------------------------------------------------
// Host orchestration
// ---------------------------------------------------------------------------
extern "C" void kernel_launch(void* const* d_in, const int* in_sizes, int n_in,
                              void* d_out, int out_size, void* d_ws,
                              size_t ws_size, hipStream_t stream)
{
  (void)n_in; (void)out_size; (void)ws_size;

  const float* x     = (const float*)d_in[0];   // [N,64]
  const float* pe    = (const float*)d_in[1];   // [N,4]
  const float* eattr = (const float*)d_in[2];   // [E,16]
  const int*   eidx  = (const int*)d_in[3];     // [2,E]
  const int* src = eidx;
  const int* dst = eidx + EE;
  auto P = [&](int i) { return (const float*)d_in[i]; };

  // workspace layout (floats; every region 32B-aligned)
  float* w = (float*)d_ws;
  auto alloc_f = [&](size_t n) {
    float* p = w; w += (n + 7) & ~(size_t)7; return p;
  };
  float* ea   = alloc_f((size_t)EE * CC);   // edge embeddings (persistent)
  float* hbuf = alloc_f((size_t)NN * CC);   // node state h
  float* qkvb = alloc_f((size_t)NN * 768);  // QKV f32 / concat / scratch
  float* b512 = alloc_f((size_t)NN * 512);
  float* c1   = alloc_f((size_t)NN * CC);
  float* c2   = alloc_f((size_t)NN * CC);
  float* c3   = alloc_f((size_t)NN * CC);
  float* pen  = alloc_f((size_t)NN * 4);
  float* stats  = alloc_f(2 * 768);
  float* pooled = alloc_f(GG * CC);
  float* t1 = alloc_f(GG * 128);
  float* t2 = alloc_f(GG * 64);
  half_t* ah    = (half_t*)alloc_f((size_t)NN * 256);  // N*512 halfs scratch
  half_t* qkvh  = (half_t*)alloc_f((size_t)NN * 384);  // N*768 halfs
  half_t* xh    = (half_t*)alloc_f((size_t)NN * 32);   // N*64 halfs
  half_t* eatth = (half_t*)alloc_f((size_t)EE * 8);    // E*16 halfs
  half_t* penh  = (half_t*)alloc_f((size_t)NN * 2);    // N*4 halfs
  half_t* whbase = (half_t*)alloc_f(2 * 1024 * 1024);  // 4M halfs: f16 weights

  auto cvt = [&](const float* s, half_t* d, size_t n) {
    k_cvt_h<<<2048, 256, 0, stream>>>(s, d, n);
  };
  auto gemm = [&](const half_t* Ah, int lda, const half_t* Wh,
                  const float* bi, float* D, int ldd,
                  int M, int K, int Nc, int relu) {
    dim3 grid((Nc + 63) / 64, (M + 63) / 64);
    k_gemm<<<grid, 128, 0, stream>>>(Ah, lda, Wh, Nc, bi, D, ldd, M, K, Nc,
                                     relu);
  };
  auto bn_stats_clear = [&](int C) {
    hipMemsetAsync(stats, 0, 2 * (size_t)C * sizeof(float), stream);
  };

  // ---- preconvert all weights to f16 (once per call) ----
  static const int widx[29] = {6, 8, 10, 12, 14, 16, 18, 20, 22, 24,
                               26, 28, 30, 32, 40, 42, 44, 46,
                               54, 56, 58, 60, 68, 70, 72, 74,
                               82, 84, 86};
  half_t* whp[96] = {};
  {
    half_t* cw = whbase;
    for (int j = 0; j < 29; ++j) {
      int i = widx[j];
      size_t n = (size_t)in_sizes[i];
      whp[i] = cw;
      k_cvt_h<<<1024, 256, 0, stream>>>(P(i), cw, n);
      cw += (n + 7) & ~(size_t)7;
    }
  }
  cvt(x, xh, (size_t)NN * 64);
  cvt(eattr, eatth, (size_t)EE * 16);

  // ---- preprocessing ----
  bn_stats_clear(4);
  k_add_colstats<<<dim3(1, (NN + 255) / 256), 256, 0, stream>>>(
      pe, nullptr, nullptr, stats, NN, 4);
  k_bn_apply<<<512, 256, 0, stream>>>(pe, pen, stats, P(4), P(5), NN, 4);
  cvt(pen, penh, (size_t)NN * 4);

  // node MLP: 64 -> 512(relu) -> 512(relu) -> 256 (into concat cols 0..255)
  gemm(xh, 64, whp[8], P(9), qkvb, 512, NN, 64, 512, 1);
  cvt(qkvb, ah, (size_t)NN * 512);
  gemm(ah, 512, whp[10], P(11), b512, 512, NN, 512, 512, 1);
  cvt(b512, ah, (size_t)NN * 512);
  gemm(ah, 512, whp[12], P(13), qkvb, 512, NN, 512, 256, 0);   // cols 0..255
  gemm(penh, 4, whp[6], P(7), qkvb + 256, 512, NN, 4, 256, 0); // cols 256..
  cvt(qkvb, ah, (size_t)NN * 512);
  gemm(ah, 512, whp[20], P(21), hbuf, 256, NN, 512, 256, 0);   // proj -> h

  // edge MLP (chunked): 16 -> 512(relu) -> 512(relu) -> 256
  for (int ch = 0; ch < NCHUNKS; ++ch) {
    const half_t* ein = eatth + (size_t)ch * ECHUNK * 16;
    float* eout = ea + (size_t)ch * ECHUNK * CC;
    gemm(ein, 16, whp[14], P(15), qkvb, 512, ECHUNK, 16, 512, 1);
    cvt(qkvb, ah, (size_t)ECHUNK * 512);
    gemm(ah, 512, whp[16], P(17), b512, 512, ECHUNK, 512, 512, 1);
    cvt(b512, ah, (size_t)ECHUNK * 512);
    gemm(ah, 512, whp[18], P(19), eout, 256, ECHUNK, 512, 256, 0);
  }

  // ---- GPS layers ----
  for (int l = 0; l < LL; ++l) {
    int b = 26 + l * 14;
    const float *qkvbias = P(b + 1), *outb = P(b + 3);
    const float *m1b = P(b + 5), *m2b = P(b + 7);
    const float *n1g = P(b + 8), *n1b = P(b + 9);
    const float *n2g = P(b + 10), *n2b = P(b + 11);
    const float *n3g = P(b + 12), *n3b = P(b + 13);

    // -- GINE branch --
    hipMemsetAsync(c1, 0, (size_t)NN * CC * sizeof(float), stream);
    k_gine_scatter<<<EE / 8, 256, 0, stream>>>(hbuf, ea, src, dst, c1, EE);
    k_add<<<4096, 256, 0, stream>>>(hbuf, c1, c1, (size_t)NN * CC); // h + agg
    cvt(c1, ah, (size_t)NN * CC);
    gemm(ah, CC, whp[22], P(23), c2, CC, NN, CC, CC, 1);
    cvt(c2, ah, (size_t)NN * CC);
    gemm(ah, CC, whp[24], P(25), c1, CC, NN, CC, CC, 0);
    bn_stats_clear(CC);
    k_add_colstats<<<dim3(1, (NN + 255) / 256), 256, 0, stream>>>(
        c1, hbuf, c2, stats, NN, CC);                   // c2 = gine + h
    k_bn_apply<<<4096, 256, 0, stream>>>(c2, c2, stats, n1g, n1b, NN, CC);
    // h1 = c2

    // -- attention branch --
    cvt(hbuf, ah, (size_t)NN * CC);
    gemm(ah, CC, whp[b + 0], qkvbias, qkvb, 768, NN, CC, 768, 0);
    cvt(qkvb, qkvh, (size_t)NN * 768);
    k_attn<<<dim3(NPER / 64, GG * HH), 128, 0, stream>>>(qkvh, c1);
    cvt(c1, ah, (size_t)NN * CC);
    gemm(ah, CC, whp[b + 2], outb, c3, CC, NN, CC, CC, 0);
    bn_stats_clear(CC);
    k_add_colstats<<<dim3(1, (NN + 255) / 256), 256, 0, stream>>>(
        c3, hbuf, c3, stats, NN, CC);                   // c3 = attn + h
    k_bn_apply<<<4096, 256, 0, stream>>>(c3, c3, stats, n2g, n2b, NN, CC);
    // h2 = c3

    // -- combine + FF --
    k_add<<<4096, 256, 0, stream>>>(c2, c3, c1, (size_t)NN * CC); // h1+h2
    cvt(c1, ah, (size_t)NN * CC);
    gemm(ah, CC, whp[b + 4], m1b, b512, 512, NN, CC, 512, 1);
    cvt(b512, ah, (size_t)NN * 512);
    gemm(ah, 512, whp[b + 6], m2b, c3, CC, NN, 512, CC, 0);
    bn_stats_clear(CC);
    k_add_colstats<<<dim3(1, (NN + 255) / 256), 256, 0, stream>>>(
        c1, c3, hbuf, stats, NN, CC);                   // h = out + ff
    k_bn_apply<<<4096, 256, 0, stream>>>(hbuf, hbuf, stats, n3g, n3b, NN, CC);
  }

  // ---- pool + head ----
  k_pool<<<GG, 256, 0, stream>>>(hbuf, pooled);
  cvt(pooled, ah, (size_t)GG * CC);
  gemm(ah, CC, whp[82], P(83), t1, 128, GG, CC, 128, 1);
  cvt(t1, ah, (size_t)GG * 128);
  gemm(ah, 128, whp[84], P(85), t2, 64, GG, 128, 64, 1);
  cvt(t2, ah, (size_t)GG * 64);
  gemm(ah, 64, whp[86], P(87), (float*)d_out, 1, GG, 64, 1, 0);
}
